// BiLSTM_CRF_12532714570283
// MI455X (gfx1250) — matrix operand order repaired and checked by reading, not compiled, not run on hardware
//
#include <hip/hip_runtime.h>
#include <hip/hip_bf16.h>
#include <math.h>

// Problem sizes (BiLSTM-CRF reference)
#define Bn  128
#define Tn  512
#define En  128
#define HDn 128
#define Gn  (4*HDn)   // 512 gate columns
#define Kn  20
#define BT  16        // batch tile per workgroup

typedef __attribute__((ext_vector_type(16))) _Float16 v16h;
typedef __attribute__((ext_vector_type(8)))  float    v8f;
typedef __attribute__((ext_vector_type(4)))  unsigned int u32x4;
typedef __attribute__((ext_vector_type(4)))  int      i32x4;
typedef __attribute__((ext_vector_type(8)))  int      i32x8;

#if defined(__has_builtin)
#  if __has_builtin(__builtin_amdgcn_tensor_load_to_lds) && __has_builtin(__builtin_amdgcn_s_wait_tensorcnt)
#    define USE_TDM 1
#  endif
#endif

__device__ __forceinline__ float sigmoidf_(float x) { return 1.f / (1.f + expf(-x)); }

// A-fragment, 16x32 f16 (M=lane%16; K: lanes0-15 -> {0-7,16-23}, lanes16-31 -> {8-15,24-31})
__device__ __forceinline__ v16h load_a16(const _Float16* base, int ld, int k0) {
  const int lane = threadIdx.x & 31;
  const _Float16* p = base + (lane & 15) * ld + k0 + ((lane >> 4) << 3);
  v16h a;
#pragma unroll
  for (int e = 0; e < 8; ++e) a[e] = p[e];
#pragma unroll
  for (int e = 0; e < 8; ++e) a[e + 8] = p[16 + e];
  return a;
}

// B-fragment, 32x16 f16 from W stored row-major [n][k] (so B = W^T).
// N = n0 + lane%16; K: lanes0-15 -> {0-7,16-23}, lanes16-31 -> {8-15,24-31}
__device__ __forceinline__ v16h load_b16(const _Float16* base, int ld, int n0, int k0) {
  const int lane = threadIdx.x & 31;
  const _Float16* p = base + (size_t)(n0 + (lane & 15)) * ld + k0 + ((lane >> 4) << 3);
  v16h b;
#pragma unroll
  for (int e = 0; e < 16; ++e) b[e] = p[(e & 7) + ((e >> 3) << 4)];
  return b;
}

// 1-D contiguous global->LDS copy via Tensor Data Mover: issue only (no wait).
// Descriptor: count=1, data_size=8B, 1-D tile of nhalf/4 elements.
__device__ __forceinline__ void tdm_issue(const _Float16* gsrc, _Float16* ldst, int nhalf) {
#ifdef USE_TDM
  unsigned long long ga = (unsigned long long)gsrc;
  unsigned la = (unsigned)(unsigned long long)ldst;   // low 32 bits = LDS offset
  unsigned n8 = (unsigned)nhalf >> 2;                 // 8-byte elements
  u32x4 g0;
  g0[0] = 1u;                                                // count=1, user descriptor
  g0[1] = la;                                                // lds_addr
  g0[2] = (unsigned)ga;                                      // global_addr[31:0]
  g0[3] = ((unsigned)(ga >> 32) & 0x01FFFFFFu) | (2u << 30); // addr[56:32] | type=2
  i32x8 g1;
  g1[0] = (int)(3u << 16);                    // data_size=8B, wg_mask=0
  g1[1] = (int)((n8 & 0xFFFFu) << 16);        // tensor_dim0[15:0]
  g1[2] = (int)((n8 >> 16) | (1u << 16));     // tensor_dim0[31:16] | tensor_dim1=1
  g1[3] = (int)((n8 & 0xFFFFu) << 16);        // tile_dim0
  g1[4] = 0;                                  // tile_dim1/2 unused
  g1[5] = (int)n8;                            // tensor_dim0_stride[31:0]
  g1[6] = 0;
  g1[7] = 0;
  i32x4 z4 = {0, 0, 0, 0};
#if defined(__clang_major__) && (__clang_major__ >= 23)
  i32x8 z8 = {0, 0, 0, 0, 0, 0, 0, 0};
  __builtin_amdgcn_tensor_load_to_lds(g0, g1, z4, z4, z8, 0);
#else
  __builtin_amdgcn_tensor_load_to_lds(g0, g1, z4, z4, 0);
#endif
#else
  for (int i = (int)(threadIdx.x & 31); i < nhalf; i += 32) ldst[i] = gsrc[i];
#endif
}

__device__ __forceinline__ void tdm_wait() {
#ifdef USE_TDM
  __builtin_amdgcn_s_wait_tensorcnt(0);
#endif
}

// ---------------- elementwise prep kernels ----------------

__global__ void cvt_kernel(const float* __restrict__ src, _Float16* __restrict__ dst, int n) {
  int i = blockIdx.x * 256 + threadIdx.x;
  if (i < n) dst[i] = (_Float16)src[i];
}

__global__ void embed_kernel(const int* __restrict__ sent, const float* __restrict__ embed,
                             _Float16* __restrict__ x) {
  size_t idx = (size_t)blockIdx.x * 256 + threadIdx.x;
  const size_t total = (size_t)Tn * Bn * En;
  if (idx >= total) return;
  int e = (int)(idx % En);
  size_t r = idx / En;
  int b = (int)(r % Bn);
  int t = (int)(r / Bn);
  int tok = sent[(size_t)b * Tn + t];
  x[idx] = (_Float16)embed[(size_t)tok * En + e];
}

// ---------------- persistent LSTM direction kernel ----------------
// Grid: (Bn/BT, 2 directions). Block: 256 (8 waves); wave w owns hidden slice [16w,16w+16).
// Whh fragments are register-resident per wave (loop-invariant across 512 steps);
// x_t tiles are double-buffered in LDS and prefetched one step ahead via the TDM;
// h ping-pongs in LDS; c lives in registers.
template <int D>
__global__ void __launch_bounds__(256)
lstm_kernel(const _Float16* __restrict__ x,    // (T, B, D) f16
            const _Float16* __restrict__ wih,  // (2, G, D) f16
            const _Float16* __restrict__ whh,  // (2, G, HD) f16
            const float* __restrict__ bias,    // (2, G) f32
            _Float16* __restrict__ y)          // (T, B, 2*HD) f16
{
  __shared__ _Float16 lds_x[2][BT * D];
  __shared__ _Float16 lds_h[2][BT * HDn];

  const int dir  = blockIdx.y;
  const int brow = blockIdx.x * BT;
  const int tid  = threadIdx.x;
  const int wave = tid >> 5;
  const int lane = tid & 31;
  const int lrow = lane & 15;
  const int lgrp = lane >> 4;

  for (int i = tid; i < 2 * BT * HDn; i += 256) lds_h[0][i] = (_Float16)0.f;

  // Hoist the per-wave Whh^T fragments into registers (16 frags = 128 VGPRs).
  const _Float16* whh_src = whh + (size_t)dir * Gn * HDn;
  v16h whb[4][4];
#pragma unroll
  for (int kc = 0; kc < 4; ++kc)
#pragma unroll
    for (int g = 0; g < 4; ++g)
      whb[kc][g] = load_b16(whh_src, HDn, g * HDn + wave * 16, kc * 32);

  const _Float16* wih_src = wih + (size_t)dir * Gn * D;
  float bg[4];
#pragma unroll
  for (int g = 0; g < 4; ++g) bg[g] = bias[dir * Gn + g * HDn + wave * 16 + lrow];

  float creg[8];
#pragma unroll
  for (int r = 0; r < 8; ++r) creg[r] = 0.f;

  __syncthreads();

  // Prologue: prefetch x tile for t=0.
  {
    const int tt0 = dir ? (Tn - 1) : 0;
    if (tid < 32) tdm_issue(x + ((size_t)tt0 * Bn + brow) * D, lds_x[0], BT * D);
  }

  int cur = 0, hcur = 0;
  for (int t = 0; t < Tn; ++t) {
    const int tt = dir ? (Tn - 1 - t) : t;
    if (tid < 32) tdm_wait();   // wait for x[cur]
    __syncthreads();
    if (t + 1 < Tn) {           // prefetch x[cur^1] for t+1; hidden under this step's GEMM
      const int ttn = dir ? (Tn - 2 - t) : (t + 1);
      if (tid < 32) tdm_issue(x + ((size_t)ttn * Bn + brow) * D, lds_x[cur ^ 1], BT * D);
    }

    v8f acc[4];
#pragma unroll
    for (int g = 0; g < 4; ++g)
#pragma unroll
      for (int i = 0; i < 8; ++i) acc[g][i] = 0.f;

    // z += x_t @ Wih^T  (Wih fragments loop-invariant; compiler keeps them register/L2-hot)
#pragma unroll
    for (int kc = 0; kc < D / 32; ++kc) {
      v16h a = load_a16(lds_x[cur], D, kc * 32);
#pragma unroll
      for (int g = 0; g < 4; ++g) {
        v16h bf = load_b16(wih_src, D, g * HDn + wave * 16, kc * 32);
        acc[g] = __builtin_amdgcn_wmma_f32_16x16x32_f16(false, a, false, bf,
                                                        (short)0, acc[g], false, false);
      }
    }
    // z += h @ Whh^T  (A from LDS, B from registers)
#pragma unroll
    for (int kc = 0; kc < HDn / 32; ++kc) {
      v16h a = load_a16(lds_h[hcur], HDn, kc * 32);
#pragma unroll
      for (int g = 0; g < 4; ++g)
        acc[g] = __builtin_amdgcn_wmma_f32_16x16x32_f16(false, a, false, whb[kc][g],
                                                        (short)0, acc[g], false, false);
    }

    // gate nonlinearity: lane handles (row = r + 8*lgrp, hidden j = 16*wave + lrow)
    _Float16* hn = lds_h[hcur ^ 1];
    const int j = wave * 16 + lrow;
#pragma unroll
    for (int r = 0; r < 8; ++r) {
      const int rb = r + 8 * lgrp;
      float zi = acc[0][r] + bg[0];
      float zf = acc[1][r] + bg[1];
      float zg = acc[2][r] + bg[2];
      float zo = acc[3][r] + bg[3];
      float c  = sigmoidf_(zf) * creg[r] + sigmoidf_(zi) * tanhf(zg);
      float h  = sigmoidf_(zo) * tanhf(c);
      creg[r] = c;
      hn[rb * HDn + j] = (_Float16)h;
      y[((size_t)tt * Bn + brow + rb) * (2 * HDn) + dir * HDn + j] = (_Float16)h;
    }
    __syncthreads();
    cur ^= 1;
    hcur ^= 1;
  }
}

// ---------------- emissions: (T,B,256) f16 x (20,256) f32 -> em[b][t][k] ----------------
__global__ void emis_kernel(const _Float16* __restrict__ y1, const float* __restrict__ Wout,
                            const float* __restrict__ bout, float* __restrict__ em) {
  const int t = blockIdx.x, b = blockIdx.y, k = threadIdx.x;
  if (k >= Kn) return;
  const _Float16* yp = y1 + ((size_t)t * Bn + b) * (2 * HDn);
  const float* wp = Wout + (size_t)k * (2 * HDn);
  float acc = bout[k];
#pragma unroll 4
  for (int h = 0; h < 2 * HDn; ++h) acc += (float)yp[h] * wp[h];
  em[((size_t)b * Tn + t) * Kn + k] = acc;
}

// ---------------- CRF NLL per batch row: one wave, alpha in lane registers ----------------
__global__ void crf_kernel(const float* __restrict__ em, const int* __restrict__ tags,
                           const float* __restrict__ trans, const float* __restrict__ startv,
                           const float* __restrict__ endv, float* __restrict__ part) {
  const int b = blockIdx.x;
  const int lane = threadIdx.x;
  const int* tg = tags + (size_t)b * Tn;
  const float* eb = em + (size_t)b * Tn * Kn;

  // gold-path score (fully parallel over t)
  float s = 0.f;
  for (int t = lane; t < Tn; t += 32) {
    int ct = tg[t];
    float e = eb[t * Kn + ct];
    s += e + ((t == 0) ? startv[ct] : trans[tg[t - 1] * Kn + ct]);
  }
  if (lane == 0) s += endv[tg[Tn - 1]];
#pragma unroll
  for (int off = 16; off > 0; off >>= 1) s += __shfl_xor(s, off, 32);

  // forward algorithm: lane j holds alpha[j] (j < K)
  const int j = (lane < Kn) ? lane : 0;
  float a = (lane < Kn) ? (startv[j] + eb[j]) : -3.0e38f;
  float tj[Kn];
#pragma unroll
  for (int i = 0; i < Kn; ++i) tj[i] = trans[i * Kn + j];
  for (int t = 1; t < Tn; ++t) {
    float m = -3.0e38f, ssum = 0.f;
#pragma unroll
    for (int i = 0; i < Kn; ++i) {
      float ai = __shfl(a, i, 32);
      float v = ai + tj[i];
      if (v > m) { ssum = ssum * expf(m - v) + 1.f; m = v; }
      else        ssum += expf(v - m);
    }
    float na = m + logf(ssum) + eb[t * Kn + j];
    a = (lane < Kn) ? na : -3.0e38f;
  }
  float v = (lane < Kn) ? (a + endv[j]) : -3.0e38f;
  float mx = v;
#pragma unroll
  for (int off = 16; off > 0; off >>= 1) mx = fmaxf(mx, __shfl_xor(mx, off, 32));
  float es = (lane < Kn) ? expf(v - mx) : 0.f;
#pragma unroll
  for (int off = 16; off > 0; off >>= 1) es += __shfl_xor(es, off, 32);
  float logZ = mx + logf(es);
  if (lane == 0) part[b] = logZ - s;   // NLL contribution = logZ - score
}

// deterministic fixed-order mean (no float atomics)
__global__ void final_kernel(const float* __restrict__ part, float* __restrict__ out) {
  if (threadIdx.x == 0 && blockIdx.x == 0) {
    float s = 0.f;
    for (int b = 0; b < Bn; ++b) s += part[b];
    out[0] = s / (float)Bn;
  }
}

extern "C" void kernel_launch(void* const* d_in, const int* in_sizes, int n_in,
                              void* d_out, int out_size, void* d_ws, size_t ws_size,
                              hipStream_t stream) {
  (void)in_sizes; (void)n_in; (void)out_size; (void)ws_size;
  const int*   sent   = (const int*)d_in[0];
  const int*   tags   = (const int*)d_in[1];
  /* d_in[2] = lengths: all == T in this problem, unused */
  const float* embed  = (const float*)d_in[3];
  const float* Wih0   = (const float*)d_in[4];
  const float* Whh0   = (const float*)d_in[5];
  const float* b0     = (const float*)d_in[6];
  const float* Wih1   = (const float*)d_in[7];
  const float* Whh1   = (const float*)d_in[8];
  const float* b1     = (const float*)d_in[9];
  const float* Wout   = (const float*)d_in[10];
  const float* bout   = (const float*)d_in[11];
  const float* trans  = (const float*)d_in[12];
  const float* startv = (const float*)d_in[13];
  const float* endv   = (const float*)d_in[14];

  char* ws = (char*)d_ws;
  size_t off = 0;
  auto alloc = [&](size_t bytes) { size_t o = off; off += (bytes + 255) & ~(size_t)255; return o; };
  _Float16* x0    = (_Float16*)(ws + alloc((size_t)Tn * Bn * En * 2));
  _Float16* y0    = (_Float16*)(ws + alloc((size_t)Tn * Bn * 2 * HDn * 2));
  _Float16* y1    = (_Float16*)(ws + alloc((size_t)Tn * Bn * 2 * HDn * 2));
  _Float16* wih0h = (_Float16*)(ws + alloc((size_t)2 * Gn * En * 2));
  _Float16* whh0h = (_Float16*)(ws + alloc((size_t)2 * Gn * HDn * 2));
  _Float16* wih1h = (_Float16*)(ws + alloc((size_t)2 * Gn * 2 * HDn * 2));
  _Float16* whh1h = (_Float16*)(ws + alloc((size_t)2 * Gn * HDn * 2));
  float*    em    = (float*)(ws + alloc((size_t)Bn * Tn * Kn * 4));
  float*    part  = (float*)(ws + alloc((size_t)Bn * 4));

  // fp32 -> f16 weight conversion
  const int nwih0 = 2 * Gn * En, nwhh = 2 * Gn * HDn, nwih1 = 2 * Gn * 2 * HDn;
  cvt_kernel<<<(nwih0 + 255) / 256, 256, 0, stream>>>(Wih0, wih0h, nwih0);
  cvt_kernel<<<(nwhh + 255) / 256, 256, 0, stream>>>(Whh0, whh0h, nwhh);
  cvt_kernel<<<(nwih1 + 255) / 256, 256, 0, stream>>>(Wih1, wih1h, nwih1);
  cvt_kernel<<<(nwhh + 255) / 256, 256, 0, stream>>>(Whh1, whh1h, nwhh);

  // embedding gather -> x0 (T,B,E) f16
  const size_t nx = (size_t)Tn * Bn * En;
  embed_kernel<<<(unsigned)((nx + 255) / 256), 256, 0, stream>>>(sent, embed, x0);

  // persistent BiLSTM layers (grid = batch-tiles x directions)
  lstm_kernel<En><<<dim3(Bn / BT, 2), 256, 0, stream>>>(x0, wih0h, whh0h, b0, y0);
  lstm_kernel<2 * HDn><<<dim3(Bn / BT, 2), 256, 0, stream>>>(y0, wih1h, whh1h, b1, y1);

  // emissions + CRF
  emis_kernel<<<dim3(Tn, Bn), 32, 0, stream>>>(y1, Wout, bout, em);
  crf_kernel<<<Bn, 32, 0, stream>>>(em, tags, trans, startv, endv, part);
  final_kernel<<<1, 1, 0, stream>>>(part, (float*)d_out);
}
